// GetSpatialGatingWeights_71751723647432
// MI455X (gfx1250) — matrix operand order repaired and checked by reading, not compiled, not run on hardware
//
#include <hip/hip_runtime.h>
#include <math.h>

// ---------------------------------------------------------------------------
// MaxViT-style spatial gating block (gMLP grid/block mix) for MI455X gfx1250.
// All GEMMs run on v_wmma_f32_16x16x32_bf16 (bf16 A/B, f32 accumulate).
// Shapes fixed by the reference: N=4, C=128, H=W=256, G=P=256.
// ---------------------------------------------------------------------------

typedef __attribute__((ext_vector_type(16))) __bf16 v16bf;
typedef __attribute__((ext_vector_type(8)))  __bf16 v8bf;
typedef __attribute__((ext_vector_type(8)))  float  v8f;

static constexpr int  Cc  = 128;     // channels
static constexpr int  C2  = 256;     // 2*C
static constexpr int  Hh  = 256;
static constexpr int  Ww  = 256;
static constexpr int  HW  = 65536;
static constexpr int  Gg  = 256;     // grid cells (16x16)
static constexpr int  Pp  = 256;     // positions per cell (16x16)

// ---- WMMA helpers ---------------------------------------------------------
// ISA layouts (CDNA5 ISA):
//  A (16x32, MxK): lane L: M=L%16; elems 0..7 -> K=8*(L>=16)+e ; 8..15 -> K=16+8*(L>=16)+(e-8)
//  B (32x16, KxN): lane L: N=L%16; elems 0..7 -> K=8*(L>=16)+e ; 8..15 -> K=16+8*(L>=16)+(e-8)
//  D (16x16 f32): elem r, lane L: M=r+8*(L>=16), N=L%16
__device__ __forceinline__ v8f wmma_bf16(v16bf a, v16bf b, v8f c) {
  return __builtin_amdgcn_wmma_f32_16x16x32_bf16(false, a, false, b, (short)0, c,
                                                 false, false);
}

// A fragment from a row-major bf16 row pointer (row = this lane's M row).
__device__ __forceinline__ v16bf load_afrag(const __bf16* __restrict__ row,
                                            int k0, int hi) {
  v8bf lo = *(const v8bf*)(row + k0 + 8 * hi);        // K = k0+8hi .. +7
  v8bf hh = *(const v8bf*)(row + k0 + 16 + 8 * hi);   // K = k0+16+8hi .. +7
  v16bf a;
#pragma unroll
  for (int e = 0; e < 8; ++e) { a[e] = lo[e]; a[e + 8] = hh[e]; }
  return a;
}

// B fragment: K = k0+8hi..+7 then k0+16+8hi..+7 of column-row N.
__device__ __forceinline__ v16bf load_bfrag(const __bf16* __restrict__ row,
                                            int k0, int hi) {
  v8bf lo = *(const v8bf*)(row + k0 + 8 * hi);
  v8bf hh = *(const v8bf*)(row + k0 + 16 + 8 * hi);
  v16bf b;
#pragma unroll
  for (int e = 0; e < 8; ++e) { b[e] = lo[e]; b[e + 8] = hh[e]; }
  return b;
}

__device__ __forceinline__ float gelu_exact(float v) {
  return 0.5f * v * (1.0f + erff(v * 0.70710678118654752440f));
}

// ---- LayerNorm statistics (deterministic two-stage) -----------------------
__global__ __launch_bounds__(256) void k_reduce1(const float* __restrict__ x,
                                                 float* __restrict__ part) {
  __shared__ float ss[256], sq[256];
  const int b = blockIdx.x;               // 4 * 256 blocks
  const int n = b >> 8, chunk = b & 255;
  size_t base = (size_t)n * Cc * HW + (size_t)chunk * 32768 + threadIdx.x;
  float s = 0.f, q = 0.f;
#pragma unroll 4
  for (int i = 0; i < 128; ++i) {
    float v = x[base + (size_t)i * 256];
    s += v; q += v * v;
  }
  ss[threadIdx.x] = s; sq[threadIdx.x] = q;
  __syncthreads();
  for (int off = 128; off > 0; off >>= 1) {
    if ((int)threadIdx.x < off) {
      ss[threadIdx.x] += ss[threadIdx.x + off];
      sq[threadIdx.x] += sq[threadIdx.x + off];
    }
    __syncthreads();
  }
  if (threadIdx.x == 0) { part[b * 2] = ss[0]; part[b * 2 + 1] = sq[0]; }
}

__global__ __launch_bounds__(256) void k_reduce2(const float* __restrict__ part,
                                                 float* __restrict__ stats) {
  __shared__ float ss[256], sq[256];
  const int n = blockIdx.x;
  ss[threadIdx.x] = part[(n * 256 + threadIdx.x) * 2];
  sq[threadIdx.x] = part[(n * 256 + threadIdx.x) * 2 + 1];
  __syncthreads();
  for (int off = 128; off > 0; off >>= 1) {
    if ((int)threadIdx.x < off) {
      ss[threadIdx.x] += ss[threadIdx.x + off];
      sq[threadIdx.x] += sq[threadIdx.x + off];
    }
    __syncthreads();
  }
  if (threadIdx.x == 0) {
    const float cnt = (float)Cc * (float)HW;
    float m = ss[0] / cnt;
    float var = sq[0] / cnt - m * m;
    stats[n * 2] = m;
    stats[n * 2 + 1] = rsqrtf(var + 1e-5f);
  }
}

// ---- weight prep: f32 -> bf16 copies + rowsum(W1) for LN folding ----------
__global__ __launch_bounds__(256) void k_prep(
    const float* __restrict__ w1, const float* __restrict__ w2,
    const float* __restrict__ w3, const float* __restrict__ w4,
    __bf16* __restrict__ wb1, __bf16* __restrict__ wb2,
    __bf16* __restrict__ wb3, __bf16* __restrict__ wb4,
    float* __restrict__ rowsum1) {
  const int tid = blockIdx.x * blockDim.x + threadIdx.x;
  const int nt = gridDim.x * blockDim.x;
  for (int i = tid; i < C2 * Cc; i += nt) wb1[i] = (__bf16)w1[i];
  for (int i = tid; i < Gg * Gg; i += nt) wb2[i] = (__bf16)w2[i];
  for (int i = tid; i < Pp * Pp; i += nt) wb3[i] = (__bf16)w3[i];
  for (int i = tid; i < Cc * C2; i += nt) wb4[i] = (__bf16)w4[i];
  for (int o = tid; o < C2; o += nt) {
    float s = 0.f;
    for (int c = 0; c < Cc; ++c) s += w1[o * Cc + c];
    rowsum1[o] = s;
  }
}

// ---- GEMM1: xp[o,pix] = W1 . LN(x), GELU, split into uT / v ---------------
// Wave = (n, 16-pixel tile). M=pixel, N=o (16 tiles = all 256 outputs), K=c.
// A frag = x column (strided over channels, lane-coalesced over pixels).
// uT layout [n][c][pos][g] (g contiguous), v layout [n][c][g][pos].
__global__ __launch_bounds__(256) void k_gemm1(
    const float* __restrict__ x, const float* __restrict__ b1,
    const float* __restrict__ rowsum1, const __bf16* __restrict__ wb1,
    const float* __restrict__ stats, __bf16* __restrict__ ubT,
    __bf16* __restrict__ vb) {
  const int gwave = (blockIdx.x * 256 + threadIdx.x) >> 5;  // 0..16383
  const int lane = threadIdx.x & 31, li = lane & 15, hi = lane >> 4;
  const int n = gwave >> 12;
  const int p0 = (gwave & 4095) * 16;
  const float mean = stats[2 * n], rstd = stats[2 * n + 1];

  v8f acc[16];
#pragma unroll
  for (int t = 0; t < 16; ++t) acc[t] = (v8f){0, 0, 0, 0, 0, 0, 0, 0};

  const float* __restrict__ xcol = x + (size_t)n * Cc * HW + p0 + li;
#pragma unroll
  for (int ks = 0; ks < 4; ++ks) {      // K = 128
    const int k0 = ks * 32;
    v16bf a;
#pragma unroll
    for (int e = 0; e < 8; ++e)
      a[e] = (__bf16)xcol[(size_t)(k0 + 8 * hi + e) * HW];
#pragma unroll
    for (int e = 0; e < 8; ++e)
      a[e + 8] = (__bf16)xcol[(size_t)(k0 + 16 + 8 * hi + e) * HW];
#pragma unroll
    for (int t = 0; t < 16; ++t) {
      v16bf b = load_bfrag(wb1 + (size_t)(t * 16 + li) * Cc, k0, hi);
      acc[t] = wmma_bf16(a, b, acc[t]);
    }
  }

  // epilogue: LN fold + bias + exact GELU, write blocked layouts
  const int pixb = p0 + 8 * hi;              // 8 consecutive pixels per lane
  const int h = pixb >> 8, w0 = pixb & 255;
  const int g = ((h >> 4) << 4) + (w0 >> 4);
  const int pos0 = ((h & 15) << 4) + (w0 & 15);
#pragma unroll
  for (int t = 0; t < 16; ++t) {
    const int o = t * 16 + li;
    const float be = b1[o] - rstd * mean * rowsum1[o];
    if (t < 8) {  // u branch -> transposed [c][pos][g]
      __bf16* dst = ubT + (((size_t)n * Cc + o) * Pp) * Gg + g;
#pragma unroll
      for (int r = 0; r < 8; ++r) {
        float vv = gelu_exact(rstd * acc[t][r] + be);
        dst[(size_t)(pos0 + r) * Gg] = (__bf16)vv;
      }
    } else {      // v branch -> natural [c][g][pos], packed b128 store
      const int c = o - Cc;
      v8bf pk;
#pragma unroll
      for (int r = 0; r < 8; ++r)
        pk[r] = (__bf16)gelu_exact(rstd * acc[t][r] + be);
      *(v8bf*)(vb + (((size_t)n * Cc + c) * Gg + g) * Pp + pos0) = pk;
    }
  }
}

// ---- GEMM2 (grid mix, u): U2[j,pos] = sum_g W2[j,g]*U[g,pos] --------------
// Orientation M=pos, N=j, K=g. A = uT rows (g contiguous). Writes NHWC cat.
__global__ __launch_bounds__(256) void k_gemm2(
    const __bf16* __restrict__ ubT, const __bf16* __restrict__ wb2,
    const float* __restrict__ b2, __bf16* __restrict__ catb) {
  const int gwave = (blockIdx.x * 256 + threadIdx.x) >> 5;  // 0..8191
  const int lane = threadIdx.x & 31, li = lane & 15, hi = lane >> 4;
  const int pt = gwave & 15;
  const int c = (gwave >> 4) & 127;
  const int n = gwave >> 11;
  const int pos0 = pt * 16;
  const __bf16* __restrict__ Arow =
      ubT + (((size_t)n * Cc + c) * Pp + (pos0 + li)) * Gg;

  v8f acc[16];
#pragma unroll
  for (int t = 0; t < 16; ++t) acc[t] = (v8f){0, 0, 0, 0, 0, 0, 0, 0};

#pragma unroll
  for (int ks = 0; ks < 8; ++ks) {     // K = 256
    const int k0 = ks * 32;
    v16bf a = load_afrag(Arow, k0, hi);
#pragma unroll
    for (int t = 0; t < 16; ++t) {
      v16bf b = load_bfrag(wb2 + (size_t)(t * 16 + li) * Gg, k0, hi);
      acc[t] = wmma_bf16(a, b, acc[t]);
    }
  }
#pragma unroll
  for (int t = 0; t < 16; ++t) {
    const int j = t * 16 + li;
    const float bb = b2[j];
    const int hb = (j >> 4) << 4, wb = (j & 15) << 4;
#pragma unroll
    for (int r = 0; r < 8; ++r) {
      const int pos = pos0 + 8 * hi + r;
      const int h = hb + (pos >> 4), w = wb + (pos & 15);
      catb[((size_t)n * HW + h * Ww + w) * C2 + c] = (__bf16)(acc[t][r] + bb);
    }
  }
}

// ---- GEMM3 (block mix, v): V2[g,j] = sum_p V[g,p]*W3[j,p] -----------------
// Orientation M=g, N=j, K=pos. A = v rows (pos contiguous). Writes NHWC cat.
__global__ __launch_bounds__(256) void k_gemm3(
    const __bf16* __restrict__ vb, const __bf16* __restrict__ wb3,
    const float* __restrict__ b3, __bf16* __restrict__ catb) {
  const int gwave = (blockIdx.x * 256 + threadIdx.x) >> 5;  // 0..8191
  const int lane = threadIdx.x & 31, li = lane & 15, hi = lane >> 4;
  const int gt = gwave & 15;
  const int c = (gwave >> 4) & 127;
  const int n = gwave >> 11;
  const int g0 = gt * 16;
  const __bf16* __restrict__ Arow =
      vb + (((size_t)n * Cc + c) * Gg + (g0 + li)) * Pp;

  v8f acc[16];
#pragma unroll
  for (int t = 0; t < 16; ++t) acc[t] = (v8f){0, 0, 0, 0, 0, 0, 0, 0};

#pragma unroll
  for (int ks = 0; ks < 8; ++ks) {     // K = 256
    const int k0 = ks * 32;
    v16bf a = load_afrag(Arow, k0, hi);
#pragma unroll
    for (int t = 0; t < 16; ++t) {
      v16bf b = load_bfrag(wb3 + (size_t)(t * 16 + li) * Pp, k0, hi);
      acc[t] = wmma_bf16(a, b, acc[t]);
    }
  }
#pragma unroll
  for (int t = 0; t < 16; ++t) {
    const int j = t * 16 + li;
    const float bb = b3[j];
#pragma unroll
    for (int r = 0; r < 8; ++r) {
      const int g = g0 + 8 * hi + r;
      const int h = ((g >> 4) << 4) + (j >> 4);
      const int w = ((g & 15) << 4) + (j & 15);
      catb[((size_t)n * HW + h * Ww + w) * C2 + Cc + c] =
          (__bf16)(acc[t][r] + bb);
    }
  }
}

// ---- GEMM4: out[o,pix] = W4 . cat -----------------------------------------
// Orientation M=pixel, N=o (8 tiles = 128 outputs), K=c' (256, contiguous in
// NHWC cat). Contiguous f32 b128 output stores.
__global__ __launch_bounds__(256) void k_gemm4(
    const __bf16* __restrict__ catb, const __bf16* __restrict__ wb4,
    const float* __restrict__ b4, float* __restrict__ out) {
  const int gwave = (blockIdx.x * 256 + threadIdx.x) >> 5;  // 0..16383
  const int lane = threadIdx.x & 31, li = lane & 15, hi = lane >> 4;
  const int n = gwave >> 12;
  const int p0 = (gwave & 4095) * 16;
  const __bf16* __restrict__ Arow = catb + ((size_t)n * HW + p0 + li) * C2;

  v8f acc[8];
#pragma unroll
  for (int t = 0; t < 8; ++t) acc[t] = (v8f){0, 0, 0, 0, 0, 0, 0, 0};

#pragma unroll
  for (int ks = 0; ks < 8; ++ks) {     // K = 256
    const int k0 = ks * 32;
    v16bf a = load_afrag(Arow, k0, hi);
#pragma unroll
    for (int t = 0; t < 8; ++t) {
      v16bf b = load_bfrag(wb4 + (size_t)(t * 16 + li) * C2, k0, hi);
      acc[t] = wmma_bf16(a, b, acc[t]);
    }
  }
#pragma unroll
  for (int t = 0; t < 8; ++t) {
    const int o = t * 16 + li;
    const float bb = b4[o];
    float* op = out + ((size_t)n * Cc + o) * HW + p0 + 8 * hi;
    float4 v0, v1;
    v0.x = acc[t][0] + bb; v0.y = acc[t][1] + bb;
    v0.z = acc[t][2] + bb; v0.w = acc[t][3] + bb;
    v1.x = acc[t][4] + bb; v1.y = acc[t][5] + bb;
    v1.z = acc[t][6] + bb; v1.w = acc[t][7] + bb;
    *(float4*)op = v0;
    *(float4*)(op + 4) = v1;
  }
}

// ---------------------------------------------------------------------------
extern "C" void kernel_launch(void* const* d_in, const int* in_sizes, int n_in,
                              void* d_out, int out_size, void* d_ws,
                              size_t ws_size, hipStream_t stream) {
  (void)in_sizes; (void)n_in; (void)out_size; (void)ws_size;
  const float* x  = (const float*)d_in[0];
  const float* w1 = (const float*)d_in[1];
  const float* b1 = (const float*)d_in[2];
  const float* w2 = (const float*)d_in[3];
  const float* b2 = (const float*)d_in[4];
  const float* w3 = (const float*)d_in[5];
  const float* b3 = (const float*)d_in[6];
  const float* w4 = (const float*)d_in[7];
  const float* b4 = (const float*)d_in[8];
  float* out = (float*)d_out;

  char* ws = (char*)d_ws;
  float*  part    = (float*)(ws + 0);                       // 8 KB
  float*  stats   = (float*)(ws + 8192);                    // 32 B
  float*  rowsum1 = (float*)(ws + 8704);                    // 1 KB
  __bf16* wb1 = (__bf16*)(ws + 16384);                      // 64 KB
  __bf16* wb2 = (__bf16*)(ws + 16384 + 65536);              // 128 KB
  __bf16* wb3 = (__bf16*)(ws + 16384 + 65536 + 131072);     // 128 KB
  __bf16* wb4 = (__bf16*)(ws + 16384 + 65536 + 2 * 131072); // 64 KB
  const size_t BIG = 1ull << 20;
  const size_t HALF = 67108864ull;                          // 64 MB each
  __bf16* ubT  = (__bf16*)(ws + BIG);
  __bf16* vb   = (__bf16*)(ws + BIG + HALF);
  __bf16* catb = (__bf16*)(ws + BIG + 2 * HALF);            // 128 MB (NHWC, 2C)

  k_reduce1<<<1024, 256, 0, stream>>>(x, part);
  k_reduce2<<<4, 256, 0, stream>>>(part, stats);
  k_prep<<<64, 256, 0, stream>>>(w1, w2, w3, w4, wb1, wb2, wb3, wb4, rowsum1);
  k_gemm1<<<2048, 256, 0, stream>>>(x, b1, rowsum1, wb1, stats, ubT, vb);
  k_gemm2<<<1024, 256, 0, stream>>>(ubT, wb2, b2, catb);
  k_gemm3<<<1024, 256, 0, stream>>>(vb, wb3, b3, catb);
  k_gemm4<<<2048, 256, 0, stream>>>(catb, wb4, b4, out);
}
